// MultiHeadAttentionUsingSDP_3934190043319
// MI455X (gfx1250) — compile-verified, matrix operand order checked by reading, not run on hardware
//
#include <hip/hip_runtime.h>
#include <hip/hip_bf16.h>
#include <math.h>
#include <stdint.h>

// ---------------------------------------------------------------------------
// MI455X (gfx1250) wave32 implementation of the fused MHA block:
//   qkv = x @ w_qkv + b_qkv ; RoPE(q,k) ; causal flash-SDPA ; out = ctx @ w_proj + b_proj
// All matmuls: v_wmma_f32_16x16x32_f16, f16 operands staged in LDS via
// GLOBAL_LOAD_ASYNC_TO_LDS_B128 with double buffering (ASYNCcnt pipeline).
// One-time f32->f16 conversion (+weight transpose) eliminates per-tile cvt.
// ---------------------------------------------------------------------------

#define B_  2
#define S_  2048
#define D_  2048
#define H_  16
#define HD_ 128

typedef _Float16 half_t;
typedef __attribute__((ext_vector_type(16))) _Float16 v16h;
typedef __attribute__((ext_vector_type(8)))  float    v8f;

#define WMMA_F32_16x16x32_F16(A, Bv, Cv) \
  __builtin_amdgcn_wmma_f32_16x16x32_f16(false, (A), false, (Bv), (short)0, (Cv), false, false)

// Low 32 bits of a flat pointer to LDS == LDS byte offset (aperture in high bits).
__device__ __forceinline__ uint32_t lds_off32(const void* p) {
  return (uint32_t)(uintptr_t)p;
}

// Async DMA: LDS[lds_off .. +15] = MEM[gaddr .. +15]; tracked by ASYNCcnt.
__device__ __forceinline__ void async_cp_b128(uint32_t lds_off, const half_t* gaddr) {
  asm volatile("global_load_async_to_lds_b128 %0, %1, off"
               :: "v"(lds_off), "v"((uint64_t)(uintptr_t)gaddr)
               : "memory");
}
__device__ __forceinline__ void wait_async0() {
  asm volatile("s_wait_asynccnt 0x0" ::: "memory");
}

// Load one 16(row)x32(k) f16 fragment from LDS, rows `ldr` halves apart,
// k contiguous (ISA 7.12.2 16-bit A layout; B frags staged N-major reuse it).
__device__ __forceinline__ v16h lds_frag_16x32(const half_t* __restrict__ base, int ldr) {
  const int lane = threadIdx.x & 31;
  const int m  = lane & 15;
  const int kh = (lane >> 4) << 3;
  const half_t* row = base + m * ldr;
  v16h r;
#pragma unroll
  for (int v = 0; v < 8; ++v) {
    const int k0 = ((v & 4) << 2) + kh + ((v & 3) << 1);
    r[2 * v]     = row[k0];
    r[2 * v + 1] = row[k0 + 1];
  }
  return r;
}

// ---------------------------------------------------------------------------
// One-time conversions
// ---------------------------------------------------------------------------
__global__ __launch_bounds__(256)
void cvt_f32_to_f16(const float* __restrict__ in, half_t* __restrict__ out) {
  const int i = blockIdx.x * 256 + threadIdx.x;
  out[i] = (half_t)in[i];
}

// in: R x C f32 (row-major)  ->  out: C x R f16 (row-major).  R,C % 32 == 0.
__global__ __launch_bounds__(256)
void transpose_cvt_f16(const float* __restrict__ in, half_t* __restrict__ out,
                       int R, int C) {
  __shared__ float tile[32][33];
  const int c0 = blockIdx.x * 32, r0 = blockIdx.y * 32;
  const int tx = threadIdx.x & 31, ty = threadIdx.x >> 5;   // 32 x 8
#pragma unroll
  for (int i = 0; i < 32; i += 8)
    tile[ty + i][tx] = in[(size_t)(r0 + ty + i) * C + c0 + tx];
  __syncthreads();
#pragma unroll
  for (int i = 0; i < 32; i += 8)
    out[(size_t)(c0 + ty + i) * R + r0 + tx] = (half_t)tile[tx][ty + i];
}

// ---------------------------------------------------------------------------
// C = A(f16, MxK) @ BT(f16, NxK)^T + bias.  128x128x32 block tile, 8 waves,
// each wave 32x64 (2x4 WMMA accs).  LDS double-buffered, staged with
// async-to-LDS B128 DMA overlapping the WMMA work of the current tile.
// ---------------------------------------------------------------------------
template <typename OutT>
__global__ __launch_bounds__(256)
void gemm_f16_wmma(const half_t* __restrict__ A, const half_t* __restrict__ BT,
                   const float* __restrict__ bias, OutT* __restrict__ C,
                   int M, int N, int K) {
  __shared__ half_t As[2][128 * 40];   // [m][k], row stride 40 halves (80B, 16B-mult)
  __shared__ half_t Bs[2][128 * 40];   // [n][k]

  const int tid  = threadIdx.x;
  const int lane = tid & 31;
  const int wid  = tid >> 5;
  const int m0 = blockIdx.y * 128;
  const int n0 = blockIdx.x * 128;
  const int wm = (wid >> 1) * 32;
  const int wn = (wid & 1) * 64;

  // staging map: each thread owns row r = tid>>1, 16 halves at cc = (tid&1)*16
  const int sr = tid >> 1;
  const int sc = (tid & 1) * 16;
  const half_t* gA = A  + (size_t)(m0 + sr) * K + sc;
  const half_t* gB = BT + (size_t)(n0 + sr) * K + sc;

  const v8f zf = {0.f, 0.f, 0.f, 0.f, 0.f, 0.f, 0.f, 0.f};
  v8f acc[2][4];
#pragma unroll
  for (int i = 0; i < 2; ++i)
#pragma unroll
    for (int j = 0; j < 4; ++j) acc[i][j] = zf;

  // prologue: stage k0 = 0 into buffer 0
  {
    const uint32_t la = lds_off32(&As[0][sr * 40 + sc]);
    const uint32_t lb = lds_off32(&Bs[0][sr * 40 + sc]);
    async_cp_b128(la,      gA);
    async_cp_b128(la + 16, gA + 8);
    async_cp_b128(lb,      gB);
    async_cp_b128(lb + 16, gB + 8);
  }
  wait_async0();
  __syncthreads();

  for (int k0 = 0; k0 < K; k0 += 32) {
    const int cur = (k0 >> 5) & 1;
    if (k0 + 32 < K) {                       // prefetch next tile into other buffer
      const int nxt = cur ^ 1;
      const uint32_t la = lds_off32(&As[nxt][sr * 40 + sc]);
      const uint32_t lb = lds_off32(&Bs[nxt][sr * 40 + sc]);
      async_cp_b128(la,      gA + k0 + 32);
      async_cp_b128(la + 16, gA + k0 + 40);
      async_cp_b128(lb,      gB + k0 + 32);
      async_cp_b128(lb + 16, gB + k0 + 40);
    }

    v16h af[2], bf[4];
#pragma unroll
    for (int i = 0; i < 2; ++i) af[i] = lds_frag_16x32(&As[cur][(wm + i * 16) * 40], 40);
#pragma unroll
    for (int j = 0; j < 4; ++j) bf[j] = lds_frag_16x32(&Bs[cur][(wn + j * 16) * 40], 40);
#pragma unroll
    for (int i = 0; i < 2; ++i)
#pragma unroll
      for (int j = 0; j < 4; ++j)
        acc[i][j] = WMMA_F32_16x16x32_F16(af[i], bf[j], acc[i][j]);

    wait_async0();                           // next buffer's DMA landed
    __syncthreads();                         // everyone done reading cur
  }

  const int chi = (lane >> 4) << 3;
  const int nn  = lane & 15;
#pragma unroll
  for (int i = 0; i < 2; ++i)
#pragma unroll
    for (int j = 0; j < 4; ++j) {
      const int col = n0 + wn + j * 16 + nn;
      const float bv = bias[col];
#pragma unroll
      for (int p = 0; p < 8; ++p) {
        const int row = m0 + wm + i * 16 + chi + p;
        C[(size_t)row * N + col] = (OutT)(acc[i][j][p] + bv);
      }
    }
}

// ---------------------------------------------------------------------------
// RoPE, in place on q and k parts of qkv_h (B,S,3,H,HD) f16.
// ---------------------------------------------------------------------------
__global__ __launch_bounds__(256)
void rope_kernel_f16(half_t* __restrict__ qkv) {
  const int tid = blockIdx.x * 256 + threadIdx.x;         // < 2*B*S*H*64
  const int j    = tid & 63;
  const int h    = (tid >> 6) & (H_ - 1);
  const int s    = (tid >> 10) & (S_ - 1);
  const int rest = tid >> 21;
  const int part = rest & 1;                              // 0=q, 1=k
  const int b    = rest >> 1;

  half_t* base = qkv + ((size_t)(b * S_ + s) * 3 * D_) + part * D_ + h * HD_;
  const float x1 = (float)base[j];
  const float x2 = (float)base[j + 64];
  const float theta = __expf((float)j * -0.14391157f);    // 10000^(-j/64)
  float sn, cs;
  __sincosf((float)s * theta, &sn, &cs);
  base[j]      = (half_t)(x1 * cs - x2 * sn);
  base[j + 64] = (half_t)(x1 * sn + x2 * cs);
}

// ---------------------------------------------------------------------------
// Causal flash attention on f16 qkv.  Grid (S/128, H, B), 8 waves, 16 query
// rows per wave.  K tile staged via async-to-LDS DMA; V tile staged
// transposed (hd-major) on the VALU path; online softmax; P re-laid out via
// per-wave LDS scratch; ctx written f16 (B, S, H*HD).
// ---------------------------------------------------------------------------
__global__ __launch_bounds__(256)
void flash_attn_wmma(const half_t* __restrict__ qkv, half_t* __restrict__ ctx) {
  __shared__ half_t Kt[32 * 136];        // [kv][hd], stride 136 halves (272B, 16B-mult)
  __shared__ half_t Vt[128 * 40];        // [hd][kv], stride 40
  __shared__ half_t Ps[8 * 16 * 32];     // per-wave P scratch [16][32]

  const int tid  = threadIdx.x;
  const int lane = tid & 31;
  const int wid  = tid >> 5;
  const int qt = blockIdx.x, h = blockIdx.y, b = blockIdx.z;
  const int m  = lane & 15;
  const int kh = (lane >> 4) << 3;
  const int nn = lane & 15;

  // --- Q fragments, pre-scaled by 1/sqrt(HD) ---
  const int sq_lane = qt * 128 + wid * 16 + m;
  const float scale = 0.08838834764831845f;
  const half_t* qbase = qkv + ((size_t)(b * S_ + sq_lane) * 3 * D_) + h * HD_;
  v16h qf[4];
#pragma unroll
  for (int kc = 0; kc < 4; ++kc) {
#pragma unroll
    for (int v = 0; v < 8; ++v) {
      const int d0 = kc * 32 + ((v & 4) << 2) + kh + ((v & 3) << 1);
      qf[kc][2 * v]     = (half_t)((float)qbase[d0]     * scale);
      qf[kc][2 * v + 1] = (half_t)((float)qbase[d0 + 1] * scale);
    }
  }

  const v8f zf = {0.f, 0.f, 0.f, 0.f, 0.f, 0.f, 0.f, 0.f};
  v8f o[8];
#pragma unroll
  for (int oc = 0; oc < 8; ++oc) o[oc] = zf;
  float row_m[8], row_l[8];
#pragma unroll
  for (int j = 0; j < 8; ++j) { row_m[j] = -3.0e38f; row_l[j] = 0.f; }

  half_t* Pw = &Ps[wid * 512];
  // staging maps (32 kv rows x 128 hd): 8 threads/row, 16 halves/thread
  const int rkv = tid >> 3;
  const int chd = (tid & 7) * 16;
  const int nkv = (qt + 1) * 4;

  for (int t = 0; t < nkv; ++t) {
    const int kv0b = t * 32;
    __syncthreads();                       // prev iter done with Kt/Vt/Ps
    // K tile: async DMA, [kv][hd]
    {
      const half_t* gk = qkv + ((size_t)(b * S_ + kv0b + rkv) * 3 * D_) + D_ + h * HD_ + chd;
      const uint32_t lk = lds_off32(&Kt[rkv * 136 + chd]);
      async_cp_b128(lk,      gk);
      async_cp_b128(lk + 16, gk + 8);
    }
    // V tile: manual transposed staging, [hd][kv]
    {
      const half_t* gv = qkv + ((size_t)(b * S_ + kv0b + rkv) * 3 * D_) + 2 * D_ + h * HD_ + chd;
#pragma unroll
      for (int i = 0; i < 16; ++i) Vt[(chd + i) * 40 + rkv] = gv[i];
    }
    wait_async0();
    __syncthreads();

    // scores: Q(16x128) @ K^T chunk (128x32) -> two 16x16 f32 accs
    v8f c0 = zf, c1 = zf;
#pragma unroll
    for (int kc = 0; kc < 4; ++kc) {
      const v16h kb0 = lds_frag_16x32(&Kt[0 * 136 + kc * 32], 136);
      const v16h kb1 = lds_frag_16x32(&Kt[16 * 136 + kc * 32], 136);
      c0 = WMMA_F32_16x16x32_F16(qf[kc], kb0, c0);
      c1 = WMMA_F32_16x16x32_F16(qf[kc], kb1, c1);
    }

    // online softmax; each lane keeps stats for its 8 rows
#pragma unroll
    for (int j = 0; j < 8; ++j) {
      const int r  = kh + j;
      const int sq = qt * 128 + wid * 16 + r;
      float s0 = (kv0b + nn      <= sq) ? c0[j] : -3.0e38f;
      float s1 = (kv0b + 16 + nn <= sq) ? c1[j] : -3.0e38f;
      float mx = fmaxf(s0, s1);
#pragma unroll
      for (int off = 8; off >= 1; off >>= 1) mx = fmaxf(mx, __shfl_xor(mx, off));
      const float mnew = fmaxf(row_m[j], mx);
      const float corr = __expf(row_m[j] - mnew);
      const float p0 = __expf(s0 - mnew);
      const float p1 = __expf(s1 - mnew);
      float ps = p0 + p1;
#pragma unroll
      for (int off = 8; off >= 1; off >>= 1) ps += __shfl_xor(ps, off);
      row_l[j] = row_l[j] * corr + ps;
      row_m[j] = mnew;
#pragma unroll
      for (int oc = 0; oc < 8; ++oc) o[oc][j] *= corr;
      Pw[r * 32 + nn]      = (half_t)p0;
      Pw[r * 32 + 16 + nn] = (half_t)p1;
    }
    __syncthreads();                       // publish P (C-layout -> A-layout)

    // ctx += P(16x32) @ V(32x128)
    const v16h pf = lds_frag_16x32(Pw, 32);
#pragma unroll
    for (int oc = 0; oc < 8; ++oc) {
      const v16h vb = lds_frag_16x32(&Vt[(oc * 16) * 40], 40);
      o[oc] = WMMA_F32_16x16x32_F16(pf, vb, o[oc]);
    }
  }

  // normalize + write ctx f16 (B, S, H*HD)
#pragma unroll
  for (int j = 0; j < 8; ++j) {
    const float inv = 1.0f / row_l[j];
    const int sq = qt * 128 + wid * 16 + kh + j;
    half_t* cp = ctx + ((size_t)(b * S_ + sq) * D_) + h * HD_;
#pragma unroll
    for (int oc = 0; oc < 8; ++oc) cp[oc * 16 + nn] = (half_t)(o[oc][j] * inv);
  }
}

// ---------------------------------------------------------------------------
extern "C" void kernel_launch(void* const* d_in, const int* in_sizes, int n_in,
                              void* d_out, int out_size, void* d_ws, size_t ws_size,
                              hipStream_t stream) {
  (void)in_sizes; (void)n_in; (void)out_size; (void)ws_size;
  const float* x      = (const float*)d_in[0];
  const float* w_qkv  = (const float*)d_in[1];
  const float* b_qkv  = (const float*)d_in[2];
  const float* w_proj = (const float*)d_in[3];
  const float* b_proj = (const float*)d_in[4];
  float* out = (float*)d_out;

  const int M = B_ * S_;                                    // 4096
  half_t* ws      = (half_t*)d_ws;
  half_t* x_h     = ws;                                     // M x D
  half_t* wqkvT_h = x_h     + (size_t)M * D_;               // 3D x D
  half_t* wprojT_h= wqkvT_h + (size_t)3 * D_ * D_;          // D x D
  half_t* qkv_h   = wprojT_h+ (size_t)D_ * D_;              // M x 3D
  half_t* ctx_h   = qkv_h   + (size_t)M * 3 * D_;           // M x D

  // 0) one-time f16 conversions (+ weight transposes K x N -> N x K)
  cvt_f32_to_f16<<<(int)((size_t)M * D_ / 256), 256, 0, stream>>>(x, x_h);
  transpose_cvt_f16<<<dim3(3 * D_ / 32, D_ / 32), 256, 0, stream>>>(w_qkv, wqkvT_h, D_, 3 * D_);
  transpose_cvt_f16<<<dim3(D_ / 32, D_ / 32), 256, 0, stream>>>(w_proj, wprojT_h, D_, D_);

  // 1) fused QKV projection (f16 out)
  gemm_f16_wmma<half_t><<<dim3(3 * D_ / 128, M / 128), 256, 0, stream>>>(
      x_h, wqkvT_h, b_qkv, qkv_h, M, 3 * D_, D_);
  // 2) RoPE on q,k in place
  rope_kernel_f16<<<2 * B_ * S_ * H_ * (HD_ / 2) / 256, 256, 0, stream>>>(qkv_h);
  // 3) causal flash attention -> ctx (f16)
  flash_attn_wmma<<<dim3(S_ / 128, H_, B_), 256, 0, stream>>>(qkv_h, ctx_h);
  // 4) output projection (f32 out + bias)
  gemm_f16_wmma<float><<<dim3(D_ / 128, M / 128), 256, 0, stream>>>(
      ctx_h, wprojT_h, b_proj, out, M, D_, D_);
}
